// GNNStack_32770600468937
// MI455X (gfx1250) — matrix operand order repaired and checked by reading, not compiled, not run on hardware
//
#include <hip/hip_runtime.h>
#include <hip/hip_bf16.h>

// ---------------------------------------------------------------------------
// GNNStack (GATv2 x2 + LN + global_max_pool + MLP) for gfx1250 (MI455X).
// Dense node-feature GEMMs use v_wmma_f32_16x16x32_bf16 (f32 accumulate),
// fused dual-output (Wl and Wr share the streamed A operand).
// Edge phase is bandwidth-bound -> fused gather/scatter with L2-hot atomics.
// ---------------------------------------------------------------------------

typedef __attribute__((ext_vector_type(16))) __bf16 v16bf;
typedef __attribute__((ext_vector_type(4)))  __bf16 v4bf;
typedef __attribute__((ext_vector_type(8)))  float  v8f;

#define F_IN   128
#define HID    256
#define NHEAD  4
#define CHEAD  64
#define NBATCH 32
#define MLP_H  64
#define NOUT   16

// ordered int key for float atomic max (monotone map f32 -> i32)
__device__ __forceinline__ int f2ord(float f) {
    int i = __float_as_int(f);
    return (i >= 0) ? i : (i ^ 0x7fffffff);
}
__device__ __forceinline__ float ord2f(int i) {
    return __int_as_float((i >= 0) ? i : (i ^ 0x7fffffff));
}
#define NEG_KEY ((int)0x80800000u)   /* f2ord(-FLT_MAX) */

// ---------------------------------------------------------------------------
// Fill helpers (workspace must be re-initialized every call)
// ---------------------------------------------------------------------------
__global__ void fill_f32(float* p, float v, long long n) {
    long long i = (long long)blockIdx.x * blockDim.x + threadIdx.x;
    long long s = (long long)gridDim.x * blockDim.x;
    for (; i < n; i += s) p[i] = v;
}
__global__ void fill_i32(int* p, int v, long long n) {
    long long i = (long long)blockIdx.x * blockDim.x + threadIdx.x;
    long long s = (long long)gridDim.x * blockDim.x;
    for (; i < n; i += s) p[i] = v;
}

// ---------------------------------------------------------------------------
// edge_attr mean (3 columns) for self-loop fill_value='mean'
// ---------------------------------------------------------------------------
__global__ void ea_sum_kernel(const float* __restrict__ ea, float* __restrict__ out, int E) {
    int i = blockIdx.x * blockDim.x + threadIdx.x;
    int s = gridDim.x * blockDim.x;
    float s0 = 0.f, s1 = 0.f, s2 = 0.f;
    for (; i < E; i += s) {
        s0 += ea[(size_t)i * 3 + 0];
        s1 += ea[(size_t)i * 3 + 1];
        s2 += ea[(size_t)i * 3 + 2];
    }
    atomicAdd(&out[0], s0);
    atomicAdd(&out[1], s1);
    atomicAdd(&out[2], s2);
}
__global__ void ea_finalize_kernel(float* out, float invE) {
    if (threadIdx.x < 3) out[threadIdx.x] *= invE;
}

// ---------------------------------------------------------------------------
// Fused dual GEMM: YL = X@Wl + bl, YR = X@Wr + br  (X: n x K, W: K x 256).
// Workgroup = 8 waves -> 32(M) x 256(N) tile of BOTH outputs; K step 32.
// A operand is staged once and shared by both products.
// LDS tiles are stored directly in WMMA fragment order:
//   A (16x32 bf16): lane = (m&15) | ((k>>4)&1)<<4 ; j = k&15
//   B (32x16 bf16): lane slice is K-contiguous -> store B tile as [n][k]
// ---------------------------------------------------------------------------
__global__ __launch_bounds__(256)
void gemm_dual_bf16_wmma_kernel(const float* __restrict__ X,
                                const float* __restrict__ Wl,
                                const float* __restrict__ Wr,
                                const float* __restrict__ bl,
                                const float* __restrict__ br,
                                float* __restrict__ YL,
                                float* __restrict__ YR,
                                int n, int K) {
    __shared__ __align__(32) __bf16 lA[2][32][16];   //  2 KB
    __shared__ __align__(32) __bf16 lBl[256][32];    // 16 KB
    __shared__ __align__(32) __bf16 lBr[256][32];    // 16 KB

    const int tid  = threadIdx.x;
    const int lane = tid & 31;
    const int wave = tid >> 5;
    const int mb   = wave >> 2;        // 0..1 : which 16-row block
    const int nb4  = wave & 3;         // 0..3 : which 64-col block
    const int m0   = blockIdx.x * 32;

    // A-staging geometry (one float4 = 4 consecutive k per thread)
    const int am   = tid >> 3;                       // row in tile 0..31
    const int aq   = tid & 7;                        // float4 chunk; k = 4*aq
    const int arow = m0 + am;
    const int arc  = (arow < n) ? arow : (n - 1);    // clamped (branch-free)
    const bool avalid = (arow < n);
    const int al   = (am & 15) | ((aq >> 2) << 4);
    const int aj   = (aq & 3) << 2;                       // {0,4,8,12}

    v8f accL[4], accR[4];
#pragma unroll
    for (int c = 0; c < 4; ++c) {
        accL[c] = (v8f){0.f,0.f,0.f,0.f,0.f,0.f,0.f,0.f};
        accR[c] = (v8f){0.f,0.f,0.f,0.f,0.f,0.f,0.f,0.f};
    }

    for (int k0 = 0; k0 < K; k0 += 32) {
        // stage A once (shared by both products)
        {
            const float4 xv = *reinterpret_cast<const float4*>(
                X + (size_t)arc * K + k0 + 4 * aq);
            v4bf p;
            p[0] = (__bf16)(avalid ? xv.x : 0.f);
            p[1] = (__bf16)(avalid ? xv.y : 0.f);
            p[2] = (__bf16)(avalid ? xv.z : 0.f);
            p[3] = (__bf16)(avalid ? xv.w : 0.f);
            *reinterpret_cast<v4bf*>(&lA[am >> 4][al][aj]) = p;
        }
        // stage Bl and Br (thread owns column tid; coalesced across threads)
#pragma unroll
        for (int q = 0; q < 8; ++q) {
            float a0 = Wl[(size_t)(k0 + 4 * q + 0) * HID + tid];
            float a1 = Wl[(size_t)(k0 + 4 * q + 1) * HID + tid];
            float a2 = Wl[(size_t)(k0 + 4 * q + 2) * HID + tid];
            float a3 = Wl[(size_t)(k0 + 4 * q + 3) * HID + tid];
            v4bf p;
            p[0] = (__bf16)a0; p[1] = (__bf16)a1;
            p[2] = (__bf16)a2; p[3] = (__bf16)a3;
            *reinterpret_cast<v4bf*>(&lBl[tid][4 * q]) = p;
            float b0 = Wr[(size_t)(k0 + 4 * q + 0) * HID + tid];
            float b1 = Wr[(size_t)(k0 + 4 * q + 1) * HID + tid];
            float b2 = Wr[(size_t)(k0 + 4 * q + 2) * HID + tid];
            float b3 = Wr[(size_t)(k0 + 4 * q + 3) * HID + tid];
            v4bf r;
            r[0] = (__bf16)b0; r[1] = (__bf16)b1;
            r[2] = (__bf16)b2; r[3] = (__bf16)b3;
            *reinterpret_cast<v4bf*>(&lBr[tid][4 * q]) = r;
        }
        __syncthreads();

        const v16bf afrag = *reinterpret_cast<const v16bf*>(&lA[mb][lane][0]);
        const int khalf = (lane >> 4) << 4;  // 0 or 16
        // preload all 8 B fragments into distinct registers so the scheduler
        // can overlap ds_loads with the WMMA chain (partial dscnt waits)
        v16bf bfl[4], bfr[4];
#pragma unroll
        for (int c = 0; c < 4; ++c) {
            int ncol = nb4 * 64 + c * 16 + (lane & 15);
            bfl[c] = *reinterpret_cast<const v16bf*>(&lBl[ncol][khalf]);
            bfr[c] = *reinterpret_cast<const v16bf*>(&lBr[ncol][khalf]);
        }
#pragma unroll
        for (int c = 0; c < 4; ++c)
            accL[c] = __builtin_amdgcn_wmma_f32_16x16x32_bf16(
                false, afrag, false, bfl[c], (short)0, accL[c], false, false);
#pragma unroll
        for (int c = 0; c < 4; ++c)
            accR[c] = __builtin_amdgcn_wmma_f32_16x16x32_bf16(
                false, afrag, false, bfr[c], (short)0, accR[c], false, false);
        __syncthreads();
    }

    // write out: VGPR r of lane L holds M = r + 8*(L>=16), N = colbase + (L&15)
#pragma unroll
    for (int c = 0; c < 4; ++c) {
        int ncol = nb4 * 64 + c * 16 + (lane & 15);
        float bvl = bl[ncol];
        float bvr = br[ncol];
#pragma unroll
        for (int r = 0; r < 8; ++r) {
            int row = m0 + mb * 16 + r + ((lane >> 4) << 3);
            if (row < n) {
                YL[(size_t)row * HID + ncol] = accL[c][r] + bvl;
                YR[(size_t)row * HID + ncol] = accR[c][r] + bvr;
            }
        }
    }
}

// ---------------------------------------------------------------------------
// Edge pass 1: logits + segment max.  One wave per edge, 8 channels/lane.
// e = ea @ We recomputed from LDS (avoids materializing 350 MB).
// ---------------------------------------------------------------------------
__global__ __launch_bounds__(256)
void edge_logits_kernel(const float* __restrict__ XL, const float* __restrict__ XR,
                        const float* __restrict__ edge_attr,
                        const int* __restrict__ ei_src, const int* __restrict__ ei_dst,
                        const float* __restrict__ We, const float* __restrict__ att,
                        const float* __restrict__ ea_mean,
                        float* __restrict__ logits, int* __restrict__ dmaxkey,
                        int E, int N) {
    __shared__ float sWe[3 * HID];
    __shared__ float sAtt[HID];
    for (int t = threadIdx.x; t < 3 * HID; t += 256) sWe[t] = We[t];
    if (threadIdx.x < HID) sAtt[threadIdx.x] = att[threadIdx.x];
    __syncthreads();

    const int lane = threadIdx.x & 31;
    const int wave = threadIdx.x >> 5;
    const int i = blockIdx.x * 8 + wave;
    const int M = E + N;
    if (i >= M) return;

    int src, dst; float ea0, ea1, ea2;
    if (i < E) {
        src = ei_src[i]; dst = ei_dst[i];
        ea0 = edge_attr[(size_t)i * 3 + 0];
        ea1 = edge_attr[(size_t)i * 3 + 1];
        ea2 = edge_attr[(size_t)i * 3 + 2];
    } else {
        src = dst = i - E;
        ea0 = ea_mean[0]; ea1 = ea_mean[1]; ea2 = ea_mean[2];
    }

    const int h = lane >> 3;                      // head of this lane's channels
    const float* xlp = XL + (size_t)src * HID + lane * 8;
    const float* xrp = XR + (size_t)dst * HID + lane * 8;
    float part = 0.f;
#pragma unroll
    for (int c = 0; c < 8; ++c) {
        int ch = lane * 8 + c;
        float e = ea0 * sWe[ch] + ea1 * sWe[HID + ch] + ea2 * sWe[2 * HID + ch];
        float m = xlp[c] + xrp[c] + e;
        m = (m > 0.f) ? m : 0.2f * m;             // leaky_relu
        part += m * sAtt[ch];
    }
    part += __shfl_xor(part, 1, 32);
    part += __shfl_xor(part, 2, 32);
    part += __shfl_xor(part, 4, 32);
    if ((lane & 7) == 0) {
        logits[(size_t)i * NHEAD + h] = part;
        atomicMax(&dmaxkey[(size_t)dst * NHEAD + h], f2ord(part));
    }
}

// ---------------------------------------------------------------------------
// Edge pass 2 (fused): a = exp(logit - max); accum[dst] += a*xl[src]; asum += a
// ---------------------------------------------------------------------------
__global__ __launch_bounds__(256)
void edge_scatter_kernel(const float* __restrict__ XL, const float* __restrict__ logits,
                         const int* __restrict__ dmaxkey,
                         const int* __restrict__ ei_src, const int* __restrict__ ei_dst,
                         float* __restrict__ asum, float* __restrict__ accum,
                         int E, int N) {
    const int lane = threadIdx.x & 31;
    const int wave = threadIdx.x >> 5;
    const int i = blockIdx.x * 8 + wave;
    const int M = E + N;
    if (i >= M) return;

    int src, dst;
    if (i < E) { src = ei_src[i]; dst = ei_dst[i]; }
    else       { src = dst = i - E; }

    const int h = lane >> 3;
    float lg = logits[(size_t)i * NHEAD + h];
    float mx = ord2f(dmaxkey[(size_t)dst * NHEAD + h]);
    float a = __expf(lg - mx);
    if ((lane & 7) == 0) atomicAdd(&asum[(size_t)dst * NHEAD + h], a);

    const float* xlp = XL + (size_t)src * HID + lane * 8;
    float* op = accum + (size_t)dst * HID + lane * 8;
#pragma unroll
    for (int c = 0; c < 8; ++c) atomicAdd(&op[c], a * xlp[c]);
}

// ---------------------------------------------------------------------------
// Node finalize for conv0: normalize + bias + relu + layernorm -> h
// ---------------------------------------------------------------------------
__global__ __launch_bounds__(256)
void node_finalize_ln_kernel(const float* __restrict__ accum, const float* __restrict__ asum,
                             const float* __restrict__ bias,
                             const float* __restrict__ lng, const float* __restrict__ lnb,
                             float* __restrict__ hout, int N) {
    const int lane = threadIdx.x & 31;
    const int wave = threadIdx.x >> 5;
    const int i = blockIdx.x * 8 + wave;
    if (i >= N) return;
    const int h = lane >> 3;
    float inv = 1.f / asum[(size_t)i * NHEAD + h];   // self-loop guarantees > 0

    float v[8];
    float mu = 0.f;
#pragma unroll
    for (int c = 0; c < 8; ++c) {
        int ch = lane * 8 + c;
        float t = accum[(size_t)i * HID + ch] * inv + bias[ch];
        t = fmaxf(t, 0.f);                           // relu before layernorm
        v[c] = t; mu += t;
    }
#pragma unroll
    for (int m = 1; m < 32; m <<= 1) mu += __shfl_xor(mu, m, 32);
    mu *= (1.f / HID);
    float var = 0.f;
#pragma unroll
    for (int c = 0; c < 8; ++c) { float d = v[c] - mu; var += d * d; }
#pragma unroll
    for (int m = 1; m < 32; m <<= 1) var += __shfl_xor(var, m, 32);
    var *= (1.f / HID);
    float rs = rsqrtf(var + 1e-5f);
#pragma unroll
    for (int c = 0; c < 8; ++c) {
        int ch = lane * 8 + c;
        hout[(size_t)i * HID + ch] = (v[c] - mu) * rs * lng[ch] + lnb[ch];
    }
}

// ---------------------------------------------------------------------------
// Node finalize for conv1: normalize + bias -> emb (d_out), relu -> pool max
// ---------------------------------------------------------------------------
__global__ __launch_bounds__(256)
void node_finalize_pool_kernel(const float* __restrict__ accum, const float* __restrict__ asum,
                               const float* __restrict__ bias, const int* __restrict__ batch,
                               float* __restrict__ emb, int* __restrict__ poolkey, int N) {
    const int lane = threadIdx.x & 31;
    const int wave = threadIdx.x >> 5;
    const int i = blockIdx.x * 8 + wave;
    if (i >= N) return;
    const int h = lane >> 3;
    float inv = 1.f / asum[(size_t)i * NHEAD + h];
    int b = batch[i];
#pragma unroll
    for (int c = 0; c < 8; ++c) {
        int ch = lane * 8 + c;
        float v = accum[(size_t)i * HID + ch] * inv + bias[ch];
        emb[(size_t)i * HID + ch] = v;
        atomicMax(&poolkey[(size_t)b * HID + ch], f2ord(fmaxf(v, 0.f)));
    }
}

// ---------------------------------------------------------------------------
// Pool decode + 3-layer MLP + log_softmax in one block (tiny: 32x256 input)
// ---------------------------------------------------------------------------
__global__ __launch_bounds__(256)
void mlp_kernel(const int* __restrict__ poolkey,
                const float* __restrict__ W1, const float* __restrict__ b1,
                const float* __restrict__ W2, const float* __restrict__ b2,
                const float* __restrict__ W3, const float* __restrict__ b3,
                float* __restrict__ out) {
    __shared__ float sg[NBATCH * HID];
    __shared__ float t1[NBATCH * MLP_H];
    __shared__ float t2[NBATCH * MLP_H];
    __shared__ float t3[NBATCH * NOUT];
    const int tid = threadIdx.x;

    for (int t = tid; t < NBATCH * HID; t += 256) sg[t] = ord2f(poolkey[t]);
    __syncthreads();

    for (int t = tid; t < NBATCH * MLP_H; t += 256) {
        int r = t >> 6, j = t & 63;
        float acc = b1[j];
        for (int k = 0; k < HID; ++k) acc += sg[r * HID + k] * W1[k * MLP_H + j];
        t1[t] = fmaxf(acc, 0.f);
    }
    __syncthreads();
    for (int t = tid; t < NBATCH * MLP_H; t += 256) {
        int r = t >> 6, j = t & 63;
        float acc = b2[j];
        for (int k = 0; k < MLP_H; ++k) acc += t1[r * MLP_H + k] * W2[k * MLP_H + j];
        t2[t] = fmaxf(acc, 0.f);
    }
    __syncthreads();
    for (int t = tid; t < NBATCH * NOUT; t += 256) {
        int r = t >> 4, j = t & 15;
        float acc = b3[j];
        for (int k = 0; k < MLP_H; ++k) acc += t2[r * MLP_H + k] * W3[k * NOUT + j];
        t3[t] = acc;
    }
    __syncthreads();
    if (tid < NBATCH) {
        float mx = -3.4e38f;
        for (int j = 0; j < NOUT; ++j) mx = fmaxf(mx, t3[tid * NOUT + j]);
        float se = 0.f;
        for (int j = 0; j < NOUT; ++j) se += __expf(t3[tid * NOUT + j] - mx);
        float ls = logf(se);
        for (int j = 0; j < NOUT; ++j) out[tid * NOUT + j] = t3[tid * NOUT + j] - mx - ls;
    }
}

// ---------------------------------------------------------------------------
// Host-side launcher
// ---------------------------------------------------------------------------
extern "C" void kernel_launch(void* const* d_in, const int* in_sizes, int n_in,
                              void* d_out, int out_size, void* d_ws, size_t ws_size,
                              hipStream_t stream) {
    const float* x    = (const float*)d_in[0];
    const float* ea   = (const float*)d_in[1];
    const float* Wl0  = (const float*)d_in[2];
    const float* bl0  = (const float*)d_in[3];
    const float* Wr0  = (const float*)d_in[4];
    const float* br0  = (const float*)d_in[5];
    const float* We0  = (const float*)d_in[6];
    const float* att0 = (const float*)d_in[7];
    const float* bi0  = (const float*)d_in[8];
    const float* Wl1  = (const float*)d_in[9];
    const float* bl1  = (const float*)d_in[10];
    const float* Wr1  = (const float*)d_in[11];
    const float* br1  = (const float*)d_in[12];
    const float* We1  = (const float*)d_in[13];
    const float* att1 = (const float*)d_in[14];
    const float* bi1  = (const float*)d_in[15];
    const float* lng  = (const float*)d_in[16];
    const float* lnb  = (const float*)d_in[17];
    const float* W1   = (const float*)d_in[18];
    const float* b1   = (const float*)d_in[19];
    const float* W2   = (const float*)d_in[20];
    const float* b2   = (const float*)d_in[21];
    const float* W3   = (const float*)d_in[22];
    const float* b3   = (const float*)d_in[23];
    const int*   ei   = (const int*)d_in[24];
    const int*   batch= (const int*)d_in[25];

    const int N = in_sizes[0] / F_IN;
    const int E = in_sizes[1] / 3;
    const int M = E + N;
    const size_t NH = (size_t)N * HID;

    const int* ei_src = ei;
    const int* ei_dst = ei + E;

    // workspace layout
    float* ws = (float*)d_ws;
    size_t o = 0;
    float* XL     = ws + o; o += NH;
    float* XR     = ws + o; o += NH;
    float* ACC    = ws + o; o += NH;
    float* HBUF   = ws + o; o += NH;
    float* LOGITS = ws + o; o += (size_t)M * NHEAD;
    float* ASUM   = ws + o; o += (size_t)N * NHEAD;
    int*   DMAX   = (int*)(ws + o); o += (size_t)N * NHEAD;
    int*   POOL   = (int*)(ws + o); o += NBATCH * HID;
    float* EAM    = ws + o; o += 4;

    float* out_emb = (float*)d_out;
    float* out_cls = out_emb + NH;

    const int eb = (M + 7) / 8;           // edge blocks (1 wave / edge)
    const int nb = (N + 7) / 8;           // node blocks
    const int gb = (N + 31) / 32;         // gemm blocks (32 rows each)

    // edge_attr mean (once)
    fill_f32<<<1, 256, 0, stream>>>(EAM, 0.f, 4);
    ea_sum_kernel<<<256, 256, 0, stream>>>(ea, EAM, E);
    ea_finalize_kernel<<<1, 64, 0, stream>>>(EAM, 1.f / (float)E);
    fill_i32<<<32, 256, 0, stream>>>(POOL, 0, NBATCH * HID);   // f2ord(0)=0, relu>=0

    // ---------------- conv0 ----------------
    fill_f32<<<2048, 256, 0, stream>>>(ACC, 0.f, (long long)NH);
    fill_f32<<<128, 256, 0, stream>>>(ASUM, 0.f, (long long)N * NHEAD);
    fill_i32<<<128, 256, 0, stream>>>(DMAX, NEG_KEY, (long long)N * NHEAD);

    gemm_dual_bf16_wmma_kernel<<<gb, 256, 0, stream>>>(x, Wl0, Wr0, bl0, br0,
                                                       XL, XR, N, F_IN);

    edge_logits_kernel<<<eb, 256, 0, stream>>>(XL, XR, ea, ei_src, ei_dst,
                                               We0, att0, EAM, LOGITS, DMAX, E, N);
    edge_scatter_kernel<<<eb, 256, 0, stream>>>(XL, LOGITS, DMAX, ei_src, ei_dst,
                                                ASUM, ACC, E, N);
    node_finalize_ln_kernel<<<nb, 256, 0, stream>>>(ACC, ASUM, bi0, lng, lnb, HBUF, N);

    // ---------------- conv1 ----------------
    fill_f32<<<2048, 256, 0, stream>>>(ACC, 0.f, (long long)NH);
    fill_f32<<<128, 256, 0, stream>>>(ASUM, 0.f, (long long)N * NHEAD);
    fill_i32<<<128, 256, 0, stream>>>(DMAX, NEG_KEY, (long long)N * NHEAD);

    gemm_dual_bf16_wmma_kernel<<<gb, 256, 0, stream>>>(HBUF, Wl1, Wr1, bl1, br1,
                                                       XL, XR, N, HID);

    edge_logits_kernel<<<eb, 256, 0, stream>>>(XL, XR, ea, ei_src, ei_dst,
                                               We1, att1, EAM, LOGITS, DMAX, E, N);
    edge_scatter_kernel<<<eb, 256, 0, stream>>>(XL, LOGITS, DMAX, ei_src, ei_dst,
                                                ASUM, ACC, E, N);
    node_finalize_pool_kernel<<<nb, 256, 0, stream>>>(ACC, ASUM, bi1, batch,
                                                      out_emb, POOL, N);

    // ---------------- readout MLP ----------------
    mlp_kernel<<<1, 256, 0, stream>>>(POOL, W1, b1, W2, b2, W3, b3, out_cls);
}
